// PhysicsProjection_78469052498676
// MI455X (gfx1250) — matrix operand order repaired and checked by reading, not compiled, not run on hardware
//
#include <hip/hip_runtime.h>
#include <hip/hip_bf16.h>

typedef __attribute__((ext_vector_type(16))) __bf16 v16bf;
typedef __attribute__((ext_vector_type(8)))  float  v8f;

#define NB        16384
#define DMODEL    256
#define XCOLS     111
#define NSLOTS    22
#define OUTCOLS   (NSLOTS*DMODEL + XCOLS)   /* 5743 */
#define LN_EPS    1e-5f
#define MTOK      32                        /* tokens per workgroup */

__constant__ int PV_BASE[17] = {0,3,6,43,46,49,61,64,67,74,77,80,92,95,98,105,108};
__constant__ int PV_SLOT[17] = {0,1,2,4,5,6,8,9,10,12,13,14,16,17,18,20,21};
__constant__ int ROT_BASE[2] = {52,83};
__constant__ int ROT_SLOT[2] = {7,15};
__constant__ int ST_BASE[2]  = {70,101};
__constant__ int ST_SLOT[2]  = {11,19};

template<int HEAD> struct Cfg;
template<> struct Cfg<0>{ static constexpr int DIN=3;  static constexpr int NTOK=17; static constexpr bool LN2=false; }; // pv
template<> struct Cfg<1>{ static constexpr int DIN=9;  static constexpr int NTOK=2;  static constexpr bool LN2=false; }; // rot
template<> struct Cfg<2>{ static constexpr int DIN=4;  static constexpr int NTOK=2;  static constexpr bool LN2=true;  }; // st
template<> struct Cfg<3>{ static constexpr int DIN=34; static constexpr int NTOK=1;  static constexpr bool LN2=true;  }; // pad

__device__ __forceinline__ float hreduce16(float v) {
  v += __shfl_xor(v, 1, 32);
  v += __shfl_xor(v, 2, 32);
  v += __shfl_xor(v, 4, 32);
  v += __shfl_xor(v, 8, 32);
  return v;   // sum across the 16 lanes of a half-wave
}

// One workgroup = 32 tokens of one head:
//   gather -> Linear(DIN->256) -> LN -> GELU(exact) -> bf16 LDS (32x256)
//   -> [32x256] @ [256x256] via v_wmma_f32_16x16x32_bf16 (each B fragment reused
//      across 2 A tiles -> 1 global b128 per WMMA) -> (+b2, optional LN) -> scatter
template<int HEAD>
__global__ __launch_bounds__(256)
void mlp_head_kernel(const float* __restrict__ x,
                     const float* __restrict__ W1, const float* __restrict__ b1,
                     const float* __restrict__ g1, const float* __restrict__ bt1,
                     const __bf16* __restrict__ W2T, const float* __restrict__ b2,
                     const float* __restrict__ g2,  const float* __restrict__ bt2,
                     float* __restrict__ out)
{
  constexpr int  DIN  = Cfg<HEAD>::DIN;
  constexpr int  NTOK = Cfg<HEAD>::NTOK;
  constexpr bool LN2  = Cfg<HEAD>::LN2;

  __shared__ float  xin[MTOK][DIN];
  __shared__ __bf16 hA[MTOK][DMODEL + 8];   // padded row stride (264 bf16) to dodge bank conflicts
  __shared__ float  accs[16][DMODEL];       // epilogue staging, reused for both 16-token halves

  const int tid = threadIdx.x;

  // ---- phase 1: gather the tiny per-token feature slices ----
  for (int idx = tid; idx < MTOK*DIN; idx += 256) {
    int m = idx / DIN, i = idx - m*DIN;
    int T  = blockIdx.x*MTOK + m;
    int bb = T / NTOK, j = T - bb*NTOK;
    int col;
    if constexpr (HEAD==0)      col = PV_BASE[j] + i;
    else if constexpr (HEAD==1) col = ROT_BASE[j] + i;
    else if constexpr (HEAD==2) col = ST_BASE[j] + i;
    else                        col = 9 + i;
    xin[m][i] = x[(size_t)bb*XCOLS + col];
  }
  __syncthreads();

  // ---- phase 2: layer1 (f32 VALU) + LayerNorm + exact GELU -> bf16 LDS ----
  for (int half = 0; half < 2; ++half) {
    const int t = half*16 + (tid >> 4), c = tid & 15;  // 16 lanes per token (half-wave)
    float vals[16];
    #pragma unroll
    for (int jj = 0; jj < 16; ++jj) {
      const int d = jj*16 + c;
      float a = b1[d];
      #pragma unroll
      for (int i = 0; i < DIN; ++i) a = fmaf(xin[t][i], W1[i*DMODEL + d], a);
      vals[jj] = a;
    }
    float s = 0.f;
    #pragma unroll
    for (int jj = 0; jj < 16; ++jj) s += vals[jj];
    const float mean = hreduce16(s) * (1.f/DMODEL);
    float vv = 0.f;
    #pragma unroll
    for (int jj = 0; jj < 16; ++jj) { float dlt = vals[jj]-mean; vv = fmaf(dlt, dlt, vv); }
    const float inv = rsqrtf(hreduce16(vv) * (1.f/DMODEL) + LN_EPS);
    #pragma unroll
    for (int jj = 0; jj < 16; ++jj) {
      const int d = jj*16 + c;
      float hn = (vals[jj]-mean)*inv*g1[d] + bt1[d];
      float ge = 0.5f*hn*(1.f + erff(hn*0.70710678118654752f));
      hA[t][d] = (__bf16)ge;
    }
  }
  __syncthreads();

  // ---- phase 3: WMMA GEMM, 8 waves x (2 M-tiles x 2 N-tiles) each, K loop of 8 ----
  const int wave = tid >> 5, lane = tid & 31;
  const int row    = lane & 15;
  const int khalfA = (lane & 16) ? 8  : 0;   // A: lanes>=16 hold K 8-15 / 24-31
  const int khalfB = (lane & 16) ? 8  : 0;   // B: lanes>=16 hold K 8-15 / 24-31
  const __bf16* w2r0 = W2T + (size_t)(wave*32 + row)*DMODEL;  // column n = wave*32 + row
  const __bf16* w2r1 = w2r0 + 16*DMODEL;                      // column n + 16
  v8f cc[2][2];
  #pragma unroll
  for (int a = 0; a < 2; ++a)
    #pragma unroll
    for (int bn = 0; bn < 2; ++bn) cc[a][bn] = (v8f)0.f;

  #pragma unroll
  for (int ks = 0; ks < 8; ++ks) {
    const int kOff = ks*32 + khalfA;
    union { v16bf v; uint4 q[2]; } a0, a1;
    a0.q[0] = *(const uint4*)&hA[row][kOff];             // tokens 0-15, K +0..7
    a0.q[1] = *(const uint4*)&hA[row][kOff + 16];        // tokens 0-15, K +16..23
    a1.q[0] = *(const uint4*)&hA[16 + row][kOff];        // tokens 16-31
    a1.q[1] = *(const uint4*)&hA[16 + row][kOff + 16];
    union { v16bf v; uint4 q[2]; } b0u, b1u;
    b0u.q[0] = *(const uint4*)(w2r0 + ks*32 + khalfB);
    b0u.q[1] = *(const uint4*)(w2r0 + ks*32 + khalfB + 16);
    b1u.q[0] = *(const uint4*)(w2r1 + ks*32 + khalfB);
    b1u.q[1] = *(const uint4*)(w2r1 + ks*32 + khalfB + 16);
    v16bf bf0 = b0u.v;
    v16bf bf1 = b1u.v;
    cc[0][0] = __builtin_amdgcn_wmma_f32_16x16x32_bf16(false, a0.v, false, bf0, (short)0, cc[0][0], false, false);
    cc[0][1] = __builtin_amdgcn_wmma_f32_16x16x32_bf16(false, a0.v, false, bf1, (short)0, cc[0][1], false, false);
    cc[1][0] = __builtin_amdgcn_wmma_f32_16x16x32_bf16(false, a1.v, false, bf0, (short)0, cc[1][0], false, false);
    cc[1][1] = __builtin_amdgcn_wmma_f32_16x16x32_bf16(false, a1.v, false, bf1, (short)0, cc[1][1], false, false);
  }

  // D layout: lane<16 -> rows 0..7, lane>=16 -> rows 8..15; n = tile_base + lane%16
  const int n0  = wave*32 + row;
  const float bb0 = b2[n0], bb1 = b2[n0 + 16];
  const int mb  = (lane & 16) ? 8 : 0;

  // ---- phase 4: per 16-token half: stage to LDS, (optional LN), scatter ----
  #pragma unroll
  for (int half = 0; half < 2; ++half) {
    __syncthreads();   // accs free (prev half consumed / first entry)
    #pragma unroll
    for (int r = 0; r < 8; ++r) {
      accs[mb + r][n0]      = cc[half][0][r] + bb0;
      accs[mb + r][n0 + 16] = cc[half][1][r] + bb1;
    }
    __syncthreads();

    const int t = tid >> 4, c = tid & 15;
    int T  = blockIdx.x*MTOK + half*16 + t;
    int bb = T / NTOK, j = T - bb*NTOK;
    int slot;
    if constexpr (HEAD==0)      slot = PV_SLOT[j];
    else if constexpr (HEAD==1) slot = ROT_SLOT[j];
    else if constexpr (HEAD==2) slot = ST_SLOT[j];
    else                        slot = 3;
    float* orow = out + (size_t)bb*OUTCOLS + slot*DMODEL;
    if constexpr (LN2) {
      float vals[16];
      #pragma unroll
      for (int jj = 0; jj < 16; ++jj) vals[jj] = accs[t][jj*16 + c];
      float s = 0.f;
      #pragma unroll
      for (int jj = 0; jj < 16; ++jj) s += vals[jj];
      const float mean = hreduce16(s) * (1.f/DMODEL);
      float vv = 0.f;
      #pragma unroll
      for (int jj = 0; jj < 16; ++jj) { float dlt = vals[jj]-mean; vv = fmaf(dlt, dlt, vv); }
      const float inv = rsqrtf(hreduce16(vv)*(1.f/DMODEL) + LN_EPS);
      #pragma unroll
      for (int jj = 0; jj < 16; ++jj) {
        const int d = jj*16 + c;
        orow[d] = (vals[jj]-mean)*inv*g2[d] + bt2[d];
      }
    } else {
      #pragma unroll
      for (int jj = 0; jj < 16; ++jj) {
        const int d = jj*16 + c;
        orow[d] = accs[t][d];
      }
    }
  }
}

// W2^T in bf16 (so B fragments are contiguous K per column row): w2t[h][n*256+k] = bf16(W2[k][n])
__global__ __launch_bounds__(256)
void prep_w2t_kernel(const float* __restrict__ pv, const float* __restrict__ rot,
                     const float* __restrict__ st, const float* __restrict__ pad,
                     __bf16* __restrict__ w2t)
{
  const int h = blockIdx.y;
  const float* W2 = (h==0) ? pv : (h==1) ? rot : (h==2) ? st : pad;
  const int idx = blockIdx.x*256 + threadIdx.x;   // 0..65535
  const int n = idx >> 8, k = idx & 255;
  w2t[(size_t)h*65536 + idx] = (__bf16)W2[k*DMODEL + n];
}

__global__ __launch_bounds__(256)
void copy_x_kernel(const float* __restrict__ x, float* __restrict__ out)
{
  int idx = blockIdx.x*256 + threadIdx.x;
  if (idx < NB*XCOLS) {
    int bb = idx / XCOLS, c = idx - bb*XCOLS;
    out[(size_t)bb*OUTCOLS + NSLOTS*DMODEL + c] = x[idx];
  }
}

extern "C" void kernel_launch(void* const* d_in, const int* in_sizes, int n_in,
                              void* d_out, int out_size, void* d_ws, size_t ws_size,
                              hipStream_t stream)
{
  (void)in_sizes; (void)n_in; (void)out_size; (void)ws_size;
  const float* x       = (const float*)d_in[0];
  const float* pv_W1   = (const float*)d_in[1];
  const float* pv_b1   = (const float*)d_in[2];
  const float* pv_g1   = (const float*)d_in[3];
  const float* pv_bt1  = (const float*)d_in[4];
  const float* pv_W2   = (const float*)d_in[5];
  const float* pv_b2   = (const float*)d_in[6];
  const float* rot_W1  = (const float*)d_in[7];
  const float* rot_b1  = (const float*)d_in[8];
  const float* rot_g1  = (const float*)d_in[9];
  const float* rot_bt1 = (const float*)d_in[10];
  const float* rot_W2  = (const float*)d_in[11];
  const float* rot_b2  = (const float*)d_in[12];
  const float* pad_W1  = (const float*)d_in[13];
  const float* pad_b1  = (const float*)d_in[14];
  const float* pad_g1  = (const float*)d_in[15];
  const float* pad_bt1 = (const float*)d_in[16];
  const float* pad_W2  = (const float*)d_in[17];
  const float* pad_b2  = (const float*)d_in[18];
  const float* pad_g2  = (const float*)d_in[19];
  const float* pad_bt2 = (const float*)d_in[20];
  const float* st_W1   = (const float*)d_in[21];
  const float* st_b1   = (const float*)d_in[22];
  const float* st_g1   = (const float*)d_in[23];
  const float* st_bt1  = (const float*)d_in[24];
  const float* st_W2   = (const float*)d_in[25];
  const float* st_b2   = (const float*)d_in[26];
  const float* st_g2   = (const float*)d_in[27];
  const float* st_bt2  = (const float*)d_in[28];

  float*  out = (float*)d_out;
  __bf16* w2t = (__bf16*)d_ws;    // 4 heads x 256x256 bf16 = 512 KB

  prep_w2t_kernel<<<dim3(256, 4), 256, 0, stream>>>(pv_W2, rot_W2, st_W2, pad_W2, w2t);

  mlp_head_kernel<0><<<(17*NB)/MTOK, 256, 0, stream>>>(x, pv_W1, pv_b1, pv_g1, pv_bt1,
                                                       w2t + 0*65536, pv_b2, nullptr, nullptr, out);
  mlp_head_kernel<1><<<( 2*NB)/MTOK, 256, 0, stream>>>(x, rot_W1, rot_b1, rot_g1, rot_bt1,
                                                       w2t + 1*65536, rot_b2, nullptr, nullptr, out);
  mlp_head_kernel<2><<<( 2*NB)/MTOK, 256, 0, stream>>>(x, st_W1, st_b1, st_g1, st_bt1,
                                                       w2t + 2*65536, st_b2, st_g2, st_bt2, out);
  mlp_head_kernel<3><<<( 1*NB)/MTOK, 256, 0, stream>>>(x, pad_W1, pad_b1, pad_g1, pad_bt1,
                                                       w2t + 3*65536, pad_b2, pad_g2, pad_bt2, out);

  copy_x_kernel<<<(NB*XCOLS + 255)/256, 256, 0, stream>>>(x, out);
}
